// Transformer_29154238005257
// MI455X (gfx1250) — matrix operand order repaired and checked by reading, not compiled, not run on hardware
//
#include <hip/hip_runtime.h>
#include <math.h>

// ---------------------------------------------------------------------------
// CDNA5 (gfx1250) transformer forward: bf16 WMMA (v_wmma_f32_16x16x32_bf16)
// for all GEMMs and attention, f32 accumulate, TDM (tensor_load_to_lds) for
// weight-tile staging with TENSORcnt-based double buffering. wave32.
// ---------------------------------------------------------------------------

typedef unsigned short u16;
typedef __attribute__((ext_vector_type(16))) __bf16 bf16x16;
typedef __attribute__((ext_vector_type(8)))  __bf16 bf16x8;
typedef __attribute__((ext_vector_type(8)))  float  f32x8;
typedef __attribute__((ext_vector_type(4)))  unsigned int uint4v;
typedef __attribute__((ext_vector_type(8)))  int int8v;
typedef __attribute__((ext_vector_type(4)))  int int4v;

#define VOCAB 32000
#define DMODEL 1024
#define NHEAD 16
#define NLAYER 4
#define BATCH 2
#define SEQ 1024
#define DK 64
#define DFF 4096
#define MROWS (BATCH*SEQ)   // 2048

// ---- helpers ---------------------------------------------------------------

__device__ __forceinline__ u16 f2bf(float f) {
  unsigned u = __float_as_uint(f);
  unsigned r = u + 0x7FFFu + ((u >> 16) & 1u);
  return (u16)(r >> 16);
}

union ABfrag { bf16x16 v; bf16x8 h[2]; };

// A-fragment (16x32, 16-bit): lane holds 16 contiguous K values [koff..koff+15]
__device__ __forceinline__ bf16x16 load_afrag(const u16* p) {
  ABfrag f;
  f.h[0] = *(const bf16x8*)(p);
  f.h[1] = *(const bf16x8*)(p + 8);
  return f.v;
}
// B-fragment (32x16, 16-bit): lane holds 16 contiguous K values of one column
__device__ __forceinline__ bf16x16 load_bfrag(const u16* p) {
  return *(const bf16x16*)(p);
}

__device__ __forceinline__ f32x8 wmma_bf16(bf16x16 a, bf16x16 b, f32x8 c) {
  return __builtin_amdgcn_wmma_f32_16x16x32_bf16(false, a, false, b,
                                                 (short)0, c, false, false);
}

// TDM: DMA a 2D bf16 tile (64 contiguous elems x 64 rows, row stride K elems)
// from global memory into LDS at lds_off. Tracked by TENSORcnt.
// amdgpu-toolchain (clang-23) 6-arg builtin form.
__device__ __forceinline__ void tdm_load_tile(const u16* gsrc, unsigned lds_off,
                                              int K) {
  unsigned long long ga = (unsigned long long)(const void*)gsrc;
  uint4v g0;
  g0[0] = 1u;                                   // count=1 (valid descriptor)
  g0[1] = lds_off;                              // lds_addr
  g0[2] = (unsigned)(ga & 0xffffffffu);         // global_addr[31:0]
  g0[3] = (unsigned)((ga >> 32) & 0x1ffffffu)   // global_addr[56:32]
          | 0x80000000u;                        // type=2 ("image")
  int8v g1;
  g1[0] = (int)(1u << 16);                      // data_size=1 (2 bytes)
  g1[1] = (int)(64u << 16);                     // tensor_dim0 = 64 (lo16)
  g1[2] = (int)(64u << 16);                     // tensor_dim1 = 64 (lo16)
  g1[3] = (int)(64u << 16);                     // tile_dim0 = 64
  g1[4] = 64;                                   // tile_dim1 = 64
  g1[5] = K;                                    // tensor_dim0_stride (lo32)
  g1[6] = 0;
  g1[7] = 0;
  int4v gz = {0, 0, 0, 0};                      // groups 2/3 unused (2D)
  int8v gz8 = {0, 0, 0, 0, 0, 0, 0, 0};
  __builtin_amdgcn_tensor_load_to_lds(g0, g1, gz, gz, gz8, 0);
}

// ---- elementwise / prep kernels -------------------------------------------

__global__ void embed_k(const int* __restrict__ idx, const float* __restrict__ wte,
                        const float* __restrict__ pe, float* __restrict__ out) {
  long long i = (long long)blockIdx.x * 256 + threadIdx.x;
  if (i >= (long long)MROWS * DMODEL) return;
  int m = (int)(i >> 10);
  int d = (int)(i & 1023);
  int s = m & (SEQ - 1);
  int tok = idx[m];
  out[i] = wte[(size_t)tok * DMODEL + d] + pe[(size_t)s * DMODEL + d];
}

// W[K,N] f32 -> Wt[N,K] bf16
__global__ void transpose_w_k(const float* __restrict__ W, u16* __restrict__ Wt,
                              long long total, int K, int N) {
  long long i = (long long)blockIdx.x * 256 + threadIdx.x;
  if (i >= total) return;
  int n = (int)(i / K);
  int k = (int)(i % K);
  Wt[i] = f2bf(W[(size_t)k * N + n]);
}

// V[b*S, D] bf16 -> Vt[b, h, dk, S] bf16
__global__ void transpose_v_k(const u16* __restrict__ V, u16* __restrict__ Vt) {
  long long i = (long long)blockIdx.x * 256 + threadIdx.x;
  if (i >= (long long)MROWS * DMODEL) return;
  int s = (int)(i & 1023);
  int d = (int)((i >> 10) & 63);
  int h = (int)((i >> 16) & 15);
  int b = (int)(i >> 20);
  Vt[i] = V[((size_t)(b * SEQ + s)) * DMODEL + h * DK + d];
}

// last-position rows of DecB -> 16x1024 padded A (rows >=2 zeroed)
__global__ void pad_last_k(const u16* __restrict__ DecB, u16* __restrict__ Apad) {
  int i = blockIdx.x * 256 + threadIdx.x;
  if (i >= 16 * DMODEL) return;
  int r = i >> 10, d = i & 1023;
  Apad[i] = (r < BATCH) ? DecB[((size_t)r * SEQ + (SEQ - 1)) * DMODEL + d] : (u16)0;
}

// ---- LayerNorm (one row per block, D=1024, 256 threads) --------------------

__global__ __launch_bounds__(256)
void layernorm_k(const float* __restrict__ x, const float* __restrict__ g,
                 const float* __restrict__ b, float* __restrict__ outF,
                 u16* __restrict__ outB) {
  __shared__ float red[256];
  int row = blockIdx.x, t = threadIdx.x;
  const float* xr = x + (size_t)row * DMODEL;
  float v[4], s = 0.f, s2 = 0.f;
#pragma unroll
  for (int i = 0; i < 4; ++i) {
    v[i] = xr[t + i * 256];
    s += v[i];
    s2 += v[i] * v[i];
  }
  red[t] = s; __syncthreads();
  for (int w = 128; w > 0; w >>= 1) { if (t < w) red[t] += red[t + w]; __syncthreads(); }
  float st = red[0]; __syncthreads();
  red[t] = s2; __syncthreads();
  for (int w = 128; w > 0; w >>= 1) { if (t < w) red[t] += red[t + w]; __syncthreads(); }
  float s2t = red[0];
  float mu = st * (1.0f / DMODEL);
  float var = s2t * (1.0f / DMODEL) - mu * mu;
  float inv = rsqrtf(var + 1e-5f);
#pragma unroll
  for (int i = 0; i < 4; ++i) {
    int d = t + i * 256;
    float y = (v[i] - mu) * inv * g[d] + b[d];
    size_t o = (size_t)row * DMODEL + d;
    if (outF) outF[o] = y;
    if (outB) outB[o] = f2bf(y);
  }
}

// ---- bf16 WMMA GEMM with TDM-staged, double-buffered weight tiles ----------
// C[M,N] = A[M,K] * Wt[N,K]^T + bias (+R). 4 waves/block; block tile 64Mx64N;
// the 64x64 Wt k-slice is staged in LDS by the Tensor Data Mover; all four
// B-fragments are fetched before the WMMA group for back-to-back matrix issue.

__global__ __launch_bounds__(128)
void gemm_bf16_k(const u16* __restrict__ A, const u16* __restrict__ Wt,
                 const float* __restrict__ bias, const float* __restrict__ R,
                 float* __restrict__ outF, u16* __restrict__ outB,
                 int Mrows, int Mvalid, int N, int K, int relu) {
  __shared__ u16 wtile[2][64 * 64];              // 2 x 8KB double buffer
  int wave = threadIdx.x >> 5, lane = threadIdx.x & 31;
  int m0 = blockIdx.y * 64 + wave * 16;
  int m0c = m0 > Mrows - 16 ? Mrows - 16 : m0;   // clamp (no early return: barriers)
  int n0 = blockIdx.x * 64;
  int lanelo = lane & 15, lanehi = lane >> 4;

  unsigned lds_b[2];
  lds_b[0] = (unsigned)(size_t)(void*)&wtile[0][0];
  lds_b[1] = (unsigned)(size_t)(void*)&wtile[1][0];
  const u16* wsrc = Wt + (size_t)n0 * K;

  f32x8 acc0 = {}, acc1 = {}, acc2 = {}, acc3 = {};
  const u16* aRow = A + (size_t)(m0c + lanelo) * K + lanehi * 16;

  int ktiles = K >> 6;                           // 64 K-elems per staged tile
  if (wave == 0) tdm_load_tile(wsrc, lds_b[0], K);   // prologue: tile 0
  bf16x16 a0 = load_afrag(aRow);
  bf16x16 a1 = load_afrag(aRow + 32);

  for (int kt = 0; kt < ktiles; ++kt) {
    if (wave == 0) __builtin_amdgcn_s_wait_tensorcnt(0);
    __syncthreads();                              // tile kt visible to all waves
    if (wave == 0 && kt + 1 < ktiles)
      tdm_load_tile(wsrc + (size_t)(kt + 1) * 64, lds_b[(kt + 1) & 1], K);

    bf16x16 an0 = a0, an1 = a1;                   // pipeline next A fragments
    if (kt + 1 < ktiles) {
      an0 = load_afrag(aRow + (kt + 1) * 64);
      an1 = load_afrag(aRow + (kt + 1) * 64 + 32);
      __builtin_prefetch(aRow + (kt + 2) * 64, 0, 3);
    }

    const u16* wb = &wtile[kt & 1][0] + lanelo * 64 + lanehi * 16;
#pragma unroll
    for (int s = 0; s < 2; ++s) {                 // two 16x16x32 substeps
      bf16x16 b0 = load_bfrag(wb + s * 32);
      bf16x16 b1 = load_bfrag(wb + s * 32 + 16 * 64);
      bf16x16 b2 = load_bfrag(wb + s * 32 + 32 * 64);
      bf16x16 b3 = load_bfrag(wb + s * 32 + 48 * 64);
      bf16x16 a = s ? a1 : a0;
      acc0 = wmma_bf16(a, b0, acc0);              // back-to-back WMMA group
      acc1 = wmma_bf16(a, b1, acc1);
      acc2 = wmma_bf16(a, b2, acc2);
      acc3 = wmma_bf16(a, b3, acc3);
    }
    a0 = an0; a1 = an1;
  }

  f32x8 accs[4] = {acc0, acc1, acc2, acc3};
#pragma unroll
  for (int nt = 0; nt < 4; ++nt) {
#pragma unroll
    for (int r = 0; r < 8; ++r) {
      int m = m0c + r + lanehi * 8;               // C layout: lanes16-31 -> M+8
      if (m >= Mvalid) continue;
      int n = n0 + nt * 16 + lanelo;
      float v = accs[nt][r];
      if (bias) v += bias[n];
      if (relu) v = v > 0.f ? v : 0.f;
      size_t o = (size_t)m * N + n;
      if (R) v += R[o];
      if (outF) outF[o] = v;
      if (outB) outB[o] = f2bf(v);
    }
  }
}

// ---- attention: one wave per (b, h, 16-query tile) -------------------------
// scores panel 16x1024 f32 (64KB) + P panel 16x1024 bf16 (32KB) in dyn LDS.
// K/V fragments software-pipelined one k-tile ahead.

__global__ __launch_bounds__(32)
void attn_k(const u16* __restrict__ Q, const u16* __restrict__ Km,
            const u16* __restrict__ Vt, u16* __restrict__ O, int causal) {
  extern __shared__ char smem[];
  float* sc = (float*)smem;                       // [16][1024]
  u16* P = (u16*)(smem + 16 * 1024 * 4);          // [16][1024]
  int lane = threadIdx.x;
  int lanelo = lane & 15, lanehi = lane >> 4;
  int qt = blockIdx.x, h = blockIdx.y, b = blockIdx.z;

  int qrow = b * SEQ + qt * 16 + lanelo;
  const u16* qbase = Q + (size_t)qrow * DMODEL + h * DK + lanehi * 16;
  bf16x16 qa0 = load_afrag(qbase);        // K dims 0..31
  bf16x16 qa1 = load_afrag(qbase + 32);   // K dims 32..63

  const u16* kp0 = Km + (size_t)(b * SEQ + lanelo) * DMODEL + h * DK + lanehi * 16;
  bf16x16 kf0 = load_bfrag(kp0);
  bf16x16 kf1 = load_bfrag(kp0 + 32);
  for (int kt = 0; kt < SEQ / 16; ++kt) {
    bf16x16 nf0 = kf0, nf1 = kf1;
    if (kt + 1 < SEQ / 16) {
      const u16* np = kp0 + (size_t)(kt + 1) * 16 * DMODEL;
      nf0 = load_bfrag(np);
      nf1 = load_bfrag(np + 32);
    }
    f32x8 acc = {};
    acc = wmma_bf16(qa0, kf0, acc);
    acc = wmma_bf16(qa1, kf1, acc);
#pragma unroll
    for (int r = 0; r < 8; ++r) {
      int qi = qt * 16 + r + lanehi * 8;
      int ki = kt * 16 + lanelo;
      float v = acc[r] * 0.125f;          // 1/sqrt(64)
      if (causal && ki > qi) v = -3.0e38f;
      sc[(r + lanehi * 8) * 1024 + ki] = v;
    }
    kf0 = nf0; kf1 = nf1;
  }
  __syncthreads();

  {  // softmax: 2 lanes per row (halves combined with shfl_xor 16)
    float* srow = sc + lanelo * 1024 + lanehi * 512;
    float mx = -3.4e38f;
    for (int e = 0; e < 512; ++e) mx = fmaxf(mx, srow[e]);
    mx = fmaxf(mx, __shfl_xor(mx, 16, 32));
    float sum = 0.f;
    for (int e = 0; e < 512; ++e) { float ex = __expf(srow[e] - mx); srow[e] = ex; sum += ex; }
    sum += __shfl_xor(sum, 16, 32);
    float inv = 1.0f / sum;
    u16* prow = P + lanelo * 1024 + lanehi * 512;
    for (int e = 0; e < 512; ++e) prow[e] = f2bf(srow[e] * inv);
  }
  __syncthreads();

  f32x8 o0 = {}, o1 = {}, o2 = {}, o3 = {};
  const u16* vbase = Vt + ((size_t)(b * NHEAD + h) * DK + lanelo) * SEQ + lanehi * 16;
  bf16x16 v0 = load_bfrag(vbase);
  bf16x16 v1 = load_bfrag(vbase + (size_t)16 * SEQ);
  bf16x16 v2 = load_bfrag(vbase + (size_t)32 * SEQ);
  bf16x16 v3 = load_bfrag(vbase + (size_t)48 * SEQ);
  for (int ks = 0; ks < SEQ / 32; ++ks) {
    bf16x16 n0 = v0, n1 = v1, n2 = v2, n3 = v3;
    if (ks + 1 < SEQ / 32) {
      const u16* vp = vbase + (ks + 1) * 32;
      n0 = load_bfrag(vp);
      n1 = load_bfrag(vp + (size_t)16 * SEQ);
      n2 = load_bfrag(vp + (size_t)32 * SEQ);
      n3 = load_bfrag(vp + (size_t)48 * SEQ);
    }
    bf16x16 pa = load_afrag(P + lanelo * 1024 + ks * 32 + lanehi * 16);
    o0 = wmma_bf16(pa, v0, o0);
    o1 = wmma_bf16(pa, v1, o1);
    o2 = wmma_bf16(pa, v2, o2);
    o3 = wmma_bf16(pa, v3, o3);
    v0 = n0; v1 = n1; v2 = n2; v3 = n3;
  }
  f32x8 oo[4] = {o0, o1, o2, o3};
#pragma unroll
  for (int nt = 0; nt < 4; ++nt)
#pragma unroll
    for (int r = 0; r < 8; ++r) {
      size_t m = (size_t)(b * SEQ + qt * 16 + r + lanehi * 8);
      O[m * DMODEL + h * DK + nt * 16 + lanelo] = f2bf(oo[nt][r]);
    }
}

// ---------------------------------------------------------------------------
// host orchestration
// ---------------------------------------------------------------------------

extern "C" void kernel_launch(void* const* d_in, const int* in_sizes, int n_in,
                              void* d_out, int out_size, void* d_ws, size_t ws_size,
                              hipStream_t stream) {
  (void)in_sizes; (void)n_in; (void)out_size; (void)ws_size;
  const int D = DMODEL;

  // --- inputs (jax tree flatten order: sorted dict keys, depth first) ---
  const int*   idx      = (const int*)  d_in[0];
  const float* dffn_b1  = (const float*)d_in[1];
  const float* dffn_b2  = (const float*)d_in[2];
  const float* dffn_w1  = (const float*)d_in[3];
  const float* dffn_w2  = (const float*)d_in[4];
  const float* dln1_b   = (const float*)d_in[5];
  const float* dln1_g   = (const float*)d_in[6];
  const float* dln2_b   = (const float*)d_in[7];
  const float* dln2_g   = (const float*)d_in[8];
  const float* dln3_b   = (const float*)d_in[9];
  const float* dln3_g   = (const float*)d_in[10];
  const float* ds_bk    = (const float*)d_in[11];
  const float* ds_bo    = (const float*)d_in[12];
  const float* ds_bq    = (const float*)d_in[13];
  const float* ds_bv    = (const float*)d_in[14];
  const float* ds_wk    = (const float*)d_in[15];
  const float* ds_wo    = (const float*)d_in[16];
  const float* ds_wq    = (const float*)d_in[17];
  const float* ds_wv    = (const float*)d_in[18];
  const float* dx_bk    = (const float*)d_in[19];
  const float* dx_bo    = (const float*)d_in[20];
  const float* dx_bq    = (const float*)d_in[21];
  const float* dx_bv    = (const float*)d_in[22];
  const float* dx_wk    = (const float*)d_in[23];
  const float* dx_wo    = (const float*)d_in[24];
  const float* dx_wq    = (const float*)d_in[25];
  const float* dx_wv    = (const float*)d_in[26];
  const float* dec_ln_b = (const float*)d_in[27];
  const float* dec_ln_g = (const float*)d_in[28];
  const float* ea_bk    = (const float*)d_in[29];
  const float* ea_bo    = (const float*)d_in[30];
  const float* ea_bq    = (const float*)d_in[31];
  const float* ea_bv    = (const float*)d_in[32];
  const float* ea_wk    = (const float*)d_in[33];
  const float* ea_wo    = (const float*)d_in[34];
  const float* ea_wq    = (const float*)d_in[35];
  const float* ea_wv    = (const float*)d_in[36];
  const float* effn_b1  = (const float*)d_in[37];
  const float* effn_b2  = (const float*)d_in[38];
  const float* effn_w1  = (const float*)d_in[39];
  const float* effn_w2  = (const float*)d_in[40];
  const float* eln1_b   = (const float*)d_in[41];
  const float* eln1_g   = (const float*)d_in[42];
  const float* eln2_b   = (const float*)d_in[43];
  const float* eln2_g   = (const float*)d_in[44];
  const float* enc_ln_b = (const float*)d_in[45];
  const float* enc_ln_g = (const float*)d_in[46];
  const float* lm_head  = (const float*)d_in[47];
  const float* pe       = (const float*)d_in[48];
  const float* wte      = (const float*)d_in[49];

  // --- workspace bump allocator ---
  char* base = (char*)d_ws;
  size_t off = 0;
  auto alloc = [&](size_t bytes) -> void* {
    void* p = base + off;
    off += (bytes + 255) & ~(size_t)255;
    return p;
  };

  auto doT = [&](const float* W, int K, int N) -> u16* {
    u16* Wt = (u16*)alloc((size_t)K * N * sizeof(u16));
    long long total = (long long)K * N;
    transpose_w_k<<<(unsigned)((total + 255) / 256), 256, 0, stream>>>(W, Wt, total, K, N);
    return Wt;
  };

  // transposed bf16 weights
  u16 *ewqT[NLAYER], *ewkT[NLAYER], *ewvT[NLAYER], *ewoT[NLAYER], *ew1T[NLAYER], *ew2T[NLAYER];
  u16 *swqT[NLAYER], *swkT[NLAYER], *swvT[NLAYER], *swoT[NLAYER];
  u16 *xwqT[NLAYER], *xwkT[NLAYER], *xwvT[NLAYER], *xwoT[NLAYER];
  u16 *dw1T[NLAYER], *dw2T[NLAYER];
  for (int l = 0; l < NLAYER; ++l) {
    size_t wl = (size_t)l * D * D;
    size_t fl1 = (size_t)l * D * DFF;
    ewqT[l] = doT(ea_wq + wl, D, D);  ewkT[l] = doT(ea_wk + wl, D, D);
    ewvT[l] = doT(ea_wv + wl, D, D);  ewoT[l] = doT(ea_wo + wl, D, D);
    ew1T[l] = doT(effn_w1 + fl1, D, DFF);
    ew2T[l] = doT(effn_w2 + fl1, DFF, D);
    swqT[l] = doT(ds_wq + wl, D, D);  swkT[l] = doT(ds_wk + wl, D, D);
    swvT[l] = doT(ds_wv + wl, D, D);  swoT[l] = doT(ds_wo + wl, D, D);
    xwqT[l] = doT(dx_wq + wl, D, D);  xwkT[l] = doT(dx_wk + wl, D, D);
    xwvT[l] = doT(dx_wv + wl, D, D);  xwoT[l] = doT(dx_wo + wl, D, D);
    dw1T[l] = doT(dffn_w1 + fl1, D, DFF);
    dw2T[l] = doT(dffn_w2 + fl1, DFF, D);
  }
  u16* lmT = doT(lm_head, D, VOCAB);

  // activations
  float* E   = (float*)alloc((size_t)MROWS * D * 4);
  float* Hd  = (float*)alloc((size_t)MROWS * D * 4);
  float* T0f = (float*)alloc((size_t)MROWS * D * 4);
  u16* T0b = (u16*)alloc((size_t)MROWS * D * 2);
  u16* Qb  = (u16*)alloc((size_t)MROWS * D * 2);
  u16* Kb  = (u16*)alloc((size_t)MROWS * D * 2);
  u16* Vb  = (u16*)alloc((size_t)MROWS * D * 2);
  u16* Vt  = (u16*)alloc((size_t)MROWS * D * 2);
  u16* Ob  = (u16*)alloc((size_t)MROWS * D * 2);
  u16* Fb  = (u16*)alloc((size_t)MROWS * DFF * 2);
  u16* EncB = (u16*)alloc((size_t)MROWS * D * 2);
  u16* DecB = (u16*)alloc((size_t)MROWS * D * 2);
  u16* Apad = (u16*)alloc((size_t)16 * D * 2);

  auto gemm = [&](const u16* A, const u16* Wt, const float* bias, const float* R,
                  float* outF, u16* outB, int Mrows, int Mvalid, int N, int K, int relu) {
    dim3 g(N / 64, (Mrows + 63) / 64);
    gemm_bf16_k<<<g, 128, 0, stream>>>(A, Wt, bias, R, outF, outB, Mrows, Mvalid, N, K, relu);
  };
  auto ln = [&](const float* x, const float* g, const float* b, float* outF, u16* outB) {
    layernorm_k<<<MROWS, 256, 0, stream>>>(x, g, b, outF, outB);
  };
  auto attn = [&](const u16* Qm, const u16* Km, const u16* Vtt, u16* Om, int causal) {
    dim3 g(SEQ / 16, NHEAD, BATCH);
    attn_k<<<g, 32, 96 * 1024, stream>>>(Qm, Km, Vtt, Om, causal);
  };
  auto vtr = [&]() {
    long long total = (long long)MROWS * D;
    transpose_v_k<<<(unsigned)((total + 255) / 256), 256, 0, stream>>>(Vb, Vt);
  };

  // --- embed (encoder and decoder both start from wte[idx] + pe) ---
  {
    long long total = (long long)MROWS * D;
    embed_k<<<(unsigned)((total + 255) / 256), 256, 0, stream>>>(idx, wte, pe, E);
    embed_k<<<(unsigned)((total + 255) / 256), 256, 0, stream>>>(idx, wte, pe, Hd);
  }

  // --- encoder ---
  for (int l = 0; l < NLAYER; ++l) {
    size_t bl = (size_t)l * D, fl = (size_t)l * DFF;
    ln(E, eln1_g + bl, eln1_b + bl, T0f, T0b);                       // h = LN(h)
    gemm(T0b, ewqT[l], ea_bq + bl, nullptr, nullptr, Qb, MROWS, MROWS, D, D, 0);
    gemm(T0b, ewkT[l], ea_bk + bl, nullptr, nullptr, Kb, MROWS, MROWS, D, D, 0);
    gemm(T0b, ewvT[l], ea_bv + bl, nullptr, nullptr, Vb, MROWS, MROWS, D, D, 0);
    vtr();
    attn(Qb, Kb, Vt, Ob, 0);
    gemm(Ob, ewoT[l], ea_bo + bl, T0f, E, nullptr, MROWS, MROWS, D, D, 0);  // h = LN(h)+attn
    ln(E, eln2_g + bl, eln2_b + bl, nullptr, T0b);
    gemm(T0b, ew1T[l], effn_b1 + fl, nullptr, nullptr, Fb, MROWS, MROWS, DFF, D, 1);
    gemm(Fb, ew2T[l], effn_b2 + bl, E, E, nullptr, MROWS, MROWS, D, DFF, 0); // h += ffn
  }
  ln(E, enc_ln_g, enc_ln_b, nullptr, EncB);                          // enc_out (bf16)

  // --- decoder ---
  for (int l = 0; l < NLAYER; ++l) {
    size_t bl = (size_t)l * D, fl = (size_t)l * DFF;
    // self-attn (causal)
    ln(Hd, dln1_g + bl, dln1_b + bl, T0f, T0b);
    gemm(T0b, swqT[l], ds_bq + bl, nullptr, nullptr, Qb, MROWS, MROWS, D, D, 0);
    gemm(T0b, swkT[l], ds_bk + bl, nullptr, nullptr, Kb, MROWS, MROWS, D, D, 0);
    gemm(T0b, swvT[l], ds_bv + bl, nullptr, nullptr, Vb, MROWS, MROWS, D, D, 0);
    vtr();
    attn(Qb, Kb, Vt, Ob, 1);
    gemm(Ob, swoT[l], ds_bo + bl, T0f, Hd, nullptr, MROWS, MROWS, D, D, 0);
    // cross-attn (K,V from enc_out)
    ln(Hd, dln2_g + bl, dln2_b + bl, T0f, T0b);
    gemm(T0b, xwqT[l], dx_bq + bl, nullptr, nullptr, Qb, MROWS, MROWS, D, D, 0);
    gemm(EncB, xwkT[l], dx_bk + bl, nullptr, nullptr, Kb, MROWS, MROWS, D, D, 0);
    gemm(EncB, xwvT[l], dx_bv + bl, nullptr, nullptr, Vb, MROWS, MROWS, D, D, 0);
    vtr();
    attn(Qb, Kb, Vt, Ob, 0);
    gemm(Ob, xwoT[l], dx_bo + bl, T0f, Hd, nullptr, MROWS, MROWS, D, D, 0);
    // ffn (residual base is ln3 output)
    ln(Hd, dln3_g + bl, dln3_b + bl, T0f, T0b);
    gemm(T0b, dw1T[l], dffn_b1 + fl, nullptr, nullptr, Fb, MROWS, MROWS, DFF, D, 1);
    gemm(Fb, dw2T[l], dffn_b2 + bl, T0f, Hd, nullptr, MROWS, MROWS, D, DFF, 0);
  }
  ln(Hd, dec_ln_g, dec_ln_b, nullptr, DecB);

  // --- lm_head on last positions only ---
  pad_last_k<<<(16 * D + 255) / 256, 256, 0, stream>>>(DecB, Apad);
  gemm(Apad, lmT, nullptr, nullptr, (float*)d_out, nullptr, 16, BATCH, VOCAB, D, 0);
}
